// RPN_Detector_KNN_30992484008030
// MI455X (gfx1250) — compile-verified, matrix operand order checked by reading, not run on hardware
//
#include <hip/hip_runtime.h>
#include <hip/hip_bf16.h>
#include <math.h>

// ---------------------------------------------------------------------------
// RPN_Detector_KNN for MI455X (gfx1250), wave32 + WMMA f32_16x16x32_f16.
// All 1x1 conv / MLP layers run as fragment-packed WMMA GEMMs with 4-wide
// output-tile blocking (X fragments reused across 4 WMMAs, W staged in LDS);
// KNN top-k runs in LDS; batch-stat BN folded into scale/shift at pack time.
// ---------------------------------------------------------------------------

typedef __attribute__((ext_vector_type(16))) _Float16 v16h;
typedef __attribute__((ext_vector_type(8)))  float    v8f;

#define BSZ   8
#define NPTS  16384
#define MNOD  512
#define BM    4096          // BSZ * MNOD
#define KF    64            // first knn k
#define KN    16            // node knn k
#define P1    262144        // BM * KF
#define P2    65536         // BM * KN
#define P3    4096          // BM

// ---- WMMA fragment index helpers (ISA 7.12.2, 16-bit A 16x32 layout) ------
__device__ __forceinline__ int klocal(int lane, int h) {
  int v = h >> 1, hf = h & 1;
  int base = (lane < 16) ? 0 : 8;
  return (v < 4) ? (base + (v << 1) + hf) : (base + 16 + ((v - 4) << 1) + hf);
}

// e = ((pt*nKt + kt)*32 + lane)*16 + h  ->  (pt, channel c, position p)
__device__ __forceinline__ void frag_decomp(int e, int nKt, int& pt, int& c, int& p) {
  int per = nKt << 9;
  pt = e / per;
  int r = e - pt * per;
  int kt = r >> 9, q = r & 511;
  int lane = q >> 4, h = q & 15;
  c = (kt << 5) + klocal(lane, h);
  p = (pt << 4) + (lane & 15);
}

// ---------------------------------------------------------------------------
// KNN 1: for each (b,m) node, top-64 nearest of 16384 points. Distances live
// in LDS as f16 (32KB), 64 rounds of parallel argmin.
// ---------------------------------------------------------------------------
__global__ void knn1_kernel(const float* __restrict__ node, const float* __restrict__ x,
                            int* __restrict__ idx) {
  __shared__ _Float16 sd[NPTS];
  __shared__ float rv[256];
  __shared__ int   ri[256];
  int bm = blockIdx.x, b = bm >> 9, m = bm & 511;
  int tid = threadIdx.x;
  const float* nb = node + (size_t)b * 3 * MNOD;
  const float* xb = x + (size_t)b * 3 * NPTS;
  float qx = nb[m], qy = nb[MNOD + m], qz = nb[2 * MNOD + m];
  for (int i = tid; i < NPTS; i += 256) {
    float dx = xb[i] - qx, dy = xb[NPTS + i] - qy, dz = xb[2 * NPTS + i] - qz;
    sd[i] = (_Float16)(dx * dx + dy * dy + dz * dz);
  }
  __syncthreads();
  for (int s = 0; s < KF; ++s) {
    float bv = 1e30f; int bi = 0;
    for (int i = tid; i < NPTS; i += 256) {
      float v = (float)sd[i];
      if (v < bv) { bv = v; bi = i; }
    }
    rv[tid] = bv; ri[tid] = bi;
    __syncthreads();
    for (int o = 128; o; o >>= 1) {
      if (tid < o && rv[tid + o] < rv[tid]) { rv[tid] = rv[tid + o]; ri[tid] = ri[tid + o]; }
      __syncthreads();
    }
    if (tid == 0) { idx[(size_t)bm * KF + s] = ri[0]; sd[ri[0]] = (_Float16)60000.0f; }
    __syncthreads();
  }
}

// KNN 2: node-to-node, top-16 of 512.
__global__ void knn2_kernel(const float* __restrict__ node, int* __restrict__ idx) {
  __shared__ float sd[MNOD];
  __shared__ float rv[128];
  __shared__ int   ri[128];
  int bm = blockIdx.x, b = bm >> 9, m = bm & 511, tid = threadIdx.x;
  const float* nb = node + (size_t)b * 3 * MNOD;
  float qx = nb[m], qy = nb[MNOD + m], qz = nb[2 * MNOD + m];
  for (int i = tid; i < MNOD; i += 128) {
    float dx = nb[i] - qx, dy = nb[MNOD + i] - qy, dz = nb[2 * MNOD + i] - qz;
    sd[i] = dx * dx + dy * dy + dz * dz;
  }
  __syncthreads();
  for (int s = 0; s < KN; ++s) {
    float bv = 1e30f; int bi = 0;
    for (int i = tid; i < MNOD; i += 128) if (sd[i] < bv) { bv = sd[i]; bi = i; }
    rv[tid] = bv; ri[tid] = bi;
    __syncthreads();
    for (int o = 64; o; o >>= 1) {
      if (tid < o && rv[tid + o] < rv[tid]) { rv[tid] = rv[tid + o]; ri[tid] = ri[tid + o]; }
      __syncthreads();
    }
    if (tid == 0) { idx[(size_t)bm * KN + s] = ri[0]; sd[ri[0]] = 1e30f; }
    __syncthreads();
  }
}

// ---------------------------------------------------------------------------
// Fragment-packed WMMA GEMM with OTB-wide output-tile blocking.
//   grid.x = (nOt/OTB) * (nPt/4), 4 waves per block, one P-tile per wave.
//   Block stages W tile-group (OTB x nKt fragments) into LDS once; each wave
//   loads one X fragment per K-step and feeds OTB WMMAs from LDS A-fragments.
// ---------------------------------------------------------------------------
template <int OTB>
__global__ void gemm_wmma_f16(const _Float16* __restrict__ Wp, const _Float16* __restrict__ Xp,
                              float* __restrict__ Y, int nPt, int nKt) {
  extern __shared__ _Float16 lw[];                 // OTB * nKt * 512 halves
  int blocksPerOtg = nPt >> 2;
  int otg = blockIdx.x / blocksPerOtg;
  int ptg = blockIdx.x - otg * blocksPerOtg;
  int wave = threadIdx.x >> 5;
  int lane = threadIdx.x & 31;
  int pt   = (ptg << 2) + wave;

  // cooperative W tile-group load: global_load_b128 -> ds_store_b128
  {
    const int4* src = (const int4*)(const void*)(Wp + (size_t)otg * OTB * nKt * 512);
    int4* dst = (int4*)(void*)lw;
    int chunks = OTB * nKt * 64;                   // 1KB per fragment = 64 int4
    for (int i = threadIdx.x; i < chunks; i += blockDim.x) dst[i] = src[i];
  }
  __syncthreads();

  const v16h* xf = (const v16h*)(const void*)Xp + (size_t)pt * nKt * 32;
  const v16h* wl = (const v16h*)(const void*)lw;
  v8f zero = {};
  v8f acc[OTB];
#pragma unroll
  for (int ob = 0; ob < OTB; ++ob) acc[ob] = zero;

  for (int kt = 0; kt < nKt; ++kt) {
    v16h b = xf[kt * 32 + lane];
    if (kt + 1 < nKt) __builtin_prefetch((const void*)(xf + (kt + 1) * 32 + lane), 0, 1);
#pragma unroll
    for (int ob = 0; ob < OTB; ++ob) {
      v16h a = wl[(ob * nKt + kt) * 32 + lane];    // ds_load from staged W
      acc[ob] = __builtin_amdgcn_wmma_f32_16x16x32_f16(false, a, false, b, (short)0,
                                                       acc[ob], false, false);
    }
  }

  // C/D layout: lane = N + 16*(M>=8), vgpr r -> M = (lane>=16?8:0)+r
  int n = lane & 15, mb = (lane >> 4) << 3;
  size_t P = (size_t)nPt << 4;
#pragma unroll
  for (int ob = 0; ob < OTB; ++ob) {
    float* yp = Y + (size_t)((otg * OTB + ob) * 16 + mb) * P + (size_t)pt * 16 + n;
#pragma unroll
    for (int r = 0; r < 8; ++r) yp[r * P] = acc[ob][r];
  }
}

// ---- weight packing into A-fragment order, zero-padded ---------------------
__global__ void pack_w_kernel(const float* __restrict__ W, _Float16* __restrict__ dst,
                              int Oreal, int Creal, int nKt, int total) {
  int e = blockIdx.x * 256 + threadIdx.x;
  if (e >= total) return;
  int per = nKt << 9;
  int ot = e / per;
  int r = e - ot * per;
  int kt = r >> 9, q = r & 511;
  int lane = q >> 4, h = q & 15;
  int o = (ot << 4) + (lane & 15);
  int c = (kt << 5) + klocal(lane, h);
  float v = (o < Oreal && c < Creal) ? W[o * Creal + c] : 0.0f;
  dst[e] = (_Float16)v;
}

// ---- BN batch statistics -> folded (scale, shift) per channel --------------
__global__ void bn_stats_kernel(const float* __restrict__ Y, const float* __restrict__ g,
                                const float* __restrict__ be, float* __restrict__ st, int P) {
  int c = blockIdx.x, tid = threadIdx.x;
  const float* row = Y + (size_t)c * P;
  float s = 0.f, s2 = 0.f;
  for (int i = tid; i < P; i += 256) { float v = row[i]; s += v; s2 += v * v; }
  __shared__ float sa[256], sb[256];
  sa[tid] = s; sb[tid] = s2;
  __syncthreads();
  for (int o = 128; o; o >>= 1) {
    if (tid < o) { sa[tid] += sa[tid + o]; sb[tid] += sb[tid + o]; }
    __syncthreads();
  }
  if (tid == 0) {
    float mean = sa[0] / P;
    float var  = sb[0] / P - mean * mean;
    float rstd = rsqrtf(var + 1e-5f);
    float sc   = g[c] * rstd;
    st[2 * c]     = sc;
    st[2 * c + 1] = be[c] - sc * mean;
  }
}

// ---- BN+ReLU -> f16 B-fragment pack ---------------------------------------
__global__ void pack_bnrelu_kernel(const float* __restrict__ Y, const float* __restrict__ st,
                                   _Float16* __restrict__ dst, int Oreal, int nKt, int nPt,
                                   int total) {
  int e = blockIdx.x * 256 + threadIdx.x;
  if (e >= total) return;
  int pt, c, p; frag_decomp(e, nKt, pt, c, p);
  float v = 0.0f;
  if (c < Oreal) {
    float y = Y[(size_t)c * ((size_t)nPt << 4) + p];
    v = fmaxf(0.0f, st[2 * c] * y + st[2 * c + 1]);
  }
  dst[e] = (_Float16)v;
}

// ---- per-(c,b,m) max over group of BN+ReLU activations --------------------
__global__ void colmax_kernel(const float* __restrict__ Y, const float* __restrict__ st,
                              float* __restrict__ outb, int O, int G) {
  int t = blockIdx.x * 256 + threadIdx.x;
  if (t >= O * BM) return;
  int c = t / BM, bm = t - c * BM;
  const float* row = Y + (size_t)c * BM * G + (size_t)bm * G;
  float sc = st[2 * c], sh = st[2 * c + 1];
  float mx = 0.0f;                                     // ReLU outputs are >= 0
  for (int k = 0; k < G; ++k) mx = fmaxf(mx, fmaxf(0.0f, sc * row[k] + sh));
  outb[t] = mx;
}

// ---- concat [bnrelu(Y), broadcast colmax] -> f16 fragments ----------------
__global__ void pack_concat_kernel(const float* __restrict__ Y, const float* __restrict__ st,
                                   const float* __restrict__ cmax, _Float16* __restrict__ dst,
                                   int Half, int nKt, int nPt, int G, int total) {
  int e = blockIdx.x * 256 + threadIdx.x;
  if (e >= total) return;
  int pt, c, p; frag_decomp(e, nKt, pt, c, p);
  size_t P = (size_t)nPt << 4;
  float v;
  if (c < Half) v = fmaxf(0.0f, st[2 * c] * Y[(size_t)c * P + p] + st[2 * c + 1]);
  else          v = cmax[(size_t)(c - Half) * BM + p / G];
  dst[e] = (_Float16)v;
}

// ---- conv1 input: gather knn, center xyz, append normals, pad to K=32 ------
__global__ void pack_x0_kernel(const float* __restrict__ x, const float* __restrict__ sn,
                               const float* __restrict__ node, const int* __restrict__ idx,
                               _Float16* __restrict__ dst, int total) {
  int e = blockIdx.x * 256 + threadIdx.x;
  if (e >= total) return;
  int pt, c, p; frag_decomp(e, 1, pt, c, p);
  int kk = p & 63, bm = p >> 6, m = bm & 511, b = bm >> 9;
  float v = 0.0f;
  if (c < 6) {
    int id = idx[(size_t)bm * KF + kk];
    if (c < 3) v = x[((size_t)b * 3 + c) * NPTS + id] - node[((size_t)b * 3 + c) * MNOD + m];
    else       v = sn[((size_t)b * 3 + (c - 3)) * NPTS + id];
  }
  dst[e] = (_Float16)v;
}

// ---- knn fusion input: [centered node coords ; gathered node_feat], pad 160
__global__ void pack_g1_kernel(const float* __restrict__ node, const float* __restrict__ nf,
                               const int* __restrict__ idx2, _Float16* __restrict__ dst,
                               int total) {
  int e = blockIdx.x * 256 + threadIdx.x;
  if (e >= total) return;
  int pt, c, p; frag_decomp(e, 5, pt, c, p);
  int kk = p & 15, bm = p >> 4, m = bm & 511, b = bm >> 9;
  float v = 0.0f;
  if (c < 131) {
    int id = idx2[(size_t)bm * KN + kk];
    if (c < 3) v = node[((size_t)b * 3 + c) * MNOD + id] - node[((size_t)b * 3 + c) * MNOD + m];
    else       v = nf[(size_t)(c - 3) * BM + (b * MNOD + id)];
  }
  dst[e] = (_Float16)v;
}

// ---- mlp head input: concat(node_feat[128], knn_feature[512]) -> K=640 -----
__global__ void pack_agg_kernel(const float* __restrict__ nf, const float* __restrict__ kf,
                                _Float16* __restrict__ dst, int total) {
  int e = blockIdx.x * 256 + threadIdx.x;
  if (e >= total) return;
  int pt, c, p; frag_decomp(e, 20, pt, c, p);
  float v = (c < 128) ? nf[(size_t)c * BM + p] : kf[(size_t)(c - 128) * BM + p];
  dst[e] = (_Float16)v;
}

// ---- head: keypoints = ks[0:3] + node ; sigmas = softplus(ks[3]) + 1e-3 ----
__global__ void final_kernel(const float* __restrict__ Ym, const float* __restrict__ bias,
                             const float* __restrict__ node, float* __restrict__ out) {
  int t = blockIdx.x * 256 + threadIdx.x;
  if (t >= BM) return;
  int m = t & 511, b = t >> 9;
  float k[4];
#pragma unroll
  for (int c = 0; c < 4; ++c) k[c] = Ym[(size_t)c * BM + t] + bias[c];
#pragma unroll
  for (int c = 0; c < 3; ++c) {
    float nd = node[((size_t)b * 3 + c) * MNOD + m];
    out[((size_t)b * 3 + c) * MNOD + m]         = nd;            // node passthrough
    out[12288 + ((size_t)b * 3 + c) * MNOD + m] = nd + k[c];     // keypoints
  }
  float s  = k[3];
  float sp = (s > 20.0f) ? s : log1pf(expf(s));
  out[24576 + t] = sp + 0.001f;                                  // sigmas
}

// ---------------------------------------------------------------------------
extern "C" void kernel_launch(void* const* d_in, const int* in_sizes, int n_in,
                              void* d_out, int out_size, void* d_ws, size_t ws_size,
                              hipStream_t stream) {
  (void)in_sizes; (void)n_in; (void)out_size; (void)ws_size;
  const float* x    = (const float*)d_in[0];
  const float* sn   = (const float*)d_in[1];
  const float* node = (const float*)d_in[2];

  // Layer table: d_in index of 'w' (g at +2, be at +3), real O / C.
  const int wIdx[13]  = {3, 7, 11, 15, 19, 23, 27, 31, 35, 39, 43, 47, 51};
  const int Oreal[13] = {64, 64, 64, 128, 128, 256, 256, 256, 512, 512, 512, 256, 4};
  const int Creal[13] = {6, 64, 64, 128, 128, 131, 256, 256, 512, 512, 640, 512, 256};

  // ---- workspace bump allocator (total ~220 MB) ----
  char* ws = (char*)d_ws;
  size_t off = 0;
  auto alloc = [&](size_t bytes) {
    size_t o = off; off += (bytes + 255) & ~(size_t)255; return o;
  };
  int*       idx1 = (int*)(ws + alloc((size_t)BM * KF * 4));
  int*       idx2 = (int*)(ws + alloc((size_t)BM * KN * 4));
  size_t woff[14]; woff[0] = 0;
  for (int i = 0; i < 13; ++i) {
    int nOt = (Oreal[i] + 15) / 16, nKt = (Creal[i] + 31) / 32;
    woff[i + 1] = woff[i] + (size_t)nOt * nKt * 512;
  }
  _Float16* wpk   = (_Float16*)(ws + alloc(woff[13] * 2));
  _Float16* xpk   = (_Float16*)(ws + alloc((size_t)33554432 * 2)); // max Kp*P fragments
  float*    Ybuf  = (float*)(ws + alloc((size_t)33554432 * 4));    // max O_pad*P
  float*    stats = (float*)(ws + alloc(1024 * 4));
  float*    tmax  = (float*)(ws + alloc((size_t)256 * BM * 4));
  float*    nodef = (float*)(ws + alloc((size_t)128 * BM * 4));
  float*    knnf  = (float*)(ws + alloc((size_t)512 * BM * 4));

  // ---- pack all weights into WMMA A-fragment order ----
  for (int i = 0; i < 13; ++i) {
    int nOt = (Oreal[i] + 15) / 16, nKt = (Creal[i] + 31) / 32;
    int total = nOt * nKt * 512;
    pack_w_kernel<<<(total + 255) / 256, 256, 0, stream>>>(
        (const float*)d_in[wIdx[i]], wpk + woff[i], Oreal[i], Creal[i], nKt, total);
  }

  auto gemmL = [&](int li, int nPt) {
    int nOt = (Oreal[li] + 15) / 16, nKt = (Creal[li] + 31) / 32;
    if ((nOt & 3) == 0) {
      int grid = (nOt >> 2) * (nPt >> 2);
      size_t sh = (size_t)4 * nKt * 512 * sizeof(_Float16);
      gemm_wmma_f16<4><<<grid, 128, sh, stream>>>(wpk + woff[li], xpk, Ybuf, nPt, nKt);
    } else {
      int grid = nOt * (nPt >> 2);
      size_t sh = (size_t)1 * nKt * 512 * sizeof(_Float16);
      gemm_wmma_f16<1><<<grid, 128, sh, stream>>>(wpk + woff[li], xpk, Ybuf, nPt, nKt);
    }
  };
  auto statsL = [&](int li, int P) {
    bn_stats_kernel<<<Oreal[li], 256, 0, stream>>>(
        Ybuf, (const float*)d_in[wIdx[li] + 2], (const float*)d_in[wIdx[li] + 3], stats, P);
  };
  auto packY = [&](int Oprev, int nKt, int nPt) {
    int total = nPt * nKt * 512;
    pack_bnrelu_kernel<<<(total + 255) / 256, 256, 0, stream>>>(Ybuf, stats, xpk, Oprev, nKt, nPt, total);
  };

  // ---- KNN searches ----
  knn1_kernel<<<BM, 256, 0, stream>>>(node, x, idx1);
  knn2_kernel<<<BM, 128, 0, stream>>>(node, idx2);

  // ---- first PointNet (positions P1 = BM*64) ----
  { int total = 16384 * 1 * 512;
    pack_x0_kernel<<<(total + 255) / 256, 256, 0, stream>>>(x, sn, node, idx1, xpk, total); }
  gemmL(0, 16384);  statsL(0, P1);  packY(64, 2, 16384);
  gemmL(1, 16384);  statsL(1, P1);  packY(64, 2, 16384);
  gemmL(2, 16384);  statsL(2, P1);
  colmax_kernel<<<(64 * BM + 255) / 256, 256, 0, stream>>>(Ybuf, stats, tmax, 64, KF);
  { int total = 16384 * 4 * 512;
    pack_concat_kernel<<<(total + 255) / 256, 256, 0, stream>>>(Ybuf, stats, tmax, xpk, 64, 4, 16384, KF, total); }
  gemmL(3, 16384);  statsL(3, P1);  packY(128, 4, 16384);
  gemmL(4, 16384);  statsL(4, P1);
  colmax_kernel<<<(128 * BM + 255) / 256, 256, 0, stream>>>(Ybuf, stats, nodef, 128, KF);

  // ---- GeneralKNNFusionModule (positions P2 = BM*16) ----
  { int total = 4096 * 5 * 512;
    pack_g1_kernel<<<(total + 255) / 256, 256, 0, stream>>>(node, nodef, idx2, xpk, total); }
  gemmL(5, 4096);   statsL(5, P2);  packY(256, 8, 4096);
  gemmL(6, 4096);   statsL(6, P2);  packY(256, 8, 4096);
  gemmL(7, 4096);   statsL(7, P2);
  colmax_kernel<<<(256 * BM + 255) / 256, 256, 0, stream>>>(Ybuf, stats, tmax, 256, KN);
  { int total = 4096 * 16 * 512;
    pack_concat_kernel<<<(total + 255) / 256, 256, 0, stream>>>(Ybuf, stats, tmax, xpk, 256, 16, 4096, KN, total); }
  gemmL(8, 4096);   statsL(8, P2);  packY(512, 16, 4096);
  gemmL(9, 4096);   statsL(9, P2);
  colmax_kernel<<<(512 * BM + 255) / 256, 256, 0, stream>>>(Ybuf, stats, knnf, 512, KN);

  // ---- MLP head (positions P3 = BM) ----
  { int total = 256 * 20 * 512;
    pack_agg_kernel<<<(total + 255) / 256, 256, 0, stream>>>(nodef, knnf, xpk, total); }
  gemmL(10, 256);   statsL(10, P3); packY(512, 16, 256);
  gemmL(11, 256);   statsL(11, P3); packY(256, 8, 256);
  gemmL(12, 256);   // mlp3: no BN, no act; bias added in final kernel
  final_kernel<<<(BM + 255) / 256, 256, 0, stream>>>(Ybuf, (const float*)d_in[52], node, (float*)d_out);
}
